// S6_66597762892257
// MI455X (gfx1250) — matrix operand order repaired and checked by reading, not compiled, not run on hardware
//
#include <hip/hip_runtime.h>

// ---------------------------------------------------------------------------
// S6 (Mamba) forward for MI455X / gfx1250, wave32 + WMMA f16->f32.
// Pipeline:
//   1. xz = x @ W_in                          (WMMA GEMM, f16 in / f32 acc)
//   2. xs = silu(causal_dwconv(xc) + b)       (elementwise)
//   3. dBC = xs @ W_x                         (WMMA GEMM, N=32)
//   4. delta_pre = d_raw @ W_dt               (WMMA GEMM, K=16 zero-padded)
//   5. selective scan over L, state-parallel  (65536 lanes, shfl_xor reduce)
//   6. out = y_gated @ W_out                  (WMMA GEMM) -> d_out
// GEMM waves compute 32x64 tile blocks (MT=2, NT=4), ping-pong pipelined in a
// single exit-free loop (acc stays in one register set; loads for buffer 0
// overlap WMMAs on buffer 1 and vice versa). __launch_bounds__(128,1) gives
// the allocator the full VGPR file (no accumulator spills).
// ---------------------------------------------------------------------------

typedef __attribute__((ext_vector_type(16))) _Float16 v16h;
typedef __attribute__((ext_vector_type(8)))  float    v8f;

#define B_DIM   2
#define L_DIM   2048
#define DMODEL  1024
#define DINNER  2048
#define DSTATE  16
#define BL      (B_DIM * L_DIM)      // 4096 rows

// ---------------- workspace layout (bytes) ----------------
static const size_t OFF_XZ  = 0;                                  // (BL, 2*DINNER) f32
static const size_t SZ_XZ   = (size_t)BL * 2 * DINNER * 4;        // 64 MB
static const size_t OFF_XS  = OFF_XZ + SZ_XZ;                     // (BL, DINNER) f32
static const size_t SZ_XS   = (size_t)BL * DINNER * 4;            // 32 MB
static const size_t OFF_DBC = OFF_XS + SZ_XS;                     // (BL, 32) f32
static const size_t SZ_DBC  = (size_t)BL * 32 * 4;
static const size_t OFF_DP  = OFF_DBC + SZ_DBC;                   // (BL, DINNER) f32
static const size_t SZ_DP   = SZ_XS;
static const size_t OFF_YG  = OFF_DP + SZ_DP;                     // (BL, DINNER) f32
static const size_t SZ_YG   = SZ_XS;
static const size_t OFF_APK = OFF_YG + SZ_YG;                     // packed A frags f16
static const size_t SZ_APK  = (size_t)BL * DINNER * 2;            // 16 MB max
static const size_t OFF_BPK = OFF_APK + SZ_APK;                   // packed B frags f16
static const size_t SZ_BPK  = (size_t)DMODEL * 2 * DINNER * 2;    // 8 MB max

// ---------------------------------------------------------------------------
// Pack A (row-major f32, leading dim lda) into WMMA 16x32 f16 A fragments.
// Fragment layout (16-bit A 16x32), contiguous k order used by this code:
//   lane<16 : row=lane,    half h -> k = h
//   lane>=16: row=lane-16, half h -> k = 16 + h
// dst index: ((mtile*Kt + ktile)*32 + lane)*16 + h.  Zero-pads row>=M, k>=K.
// ---------------------------------------------------------------------------
__global__ void pack_a_f16(const float* __restrict__ src, int lda, int M, int K,
                           int Kt, _Float16* __restrict__ dst, int total) {
  int tid = blockIdx.x * blockDim.x + threadIdx.x;
  if (tid >= total) return;
  int lane  = tid & 31;
  int kt    = (tid >> 5) % Kt;
  int mt    = tid / (Kt * 32);
  int row   = mt * 16 + (lane & 15);
  bool hiL  = lane >= 16;
  int kbase = kt * 32;
  _Float16* out = dst + ((size_t)(mt * Kt + kt) * 32 + lane) * 16;
#pragma unroll
  for (int h = 0; h < 16; ++h) {
    int k = hiL ? h + 16 : h;
    k += kbase;
    float v = (row < M && k < K) ? src[(size_t)row * lda + k] : 0.0f;
    out[h] = (_Float16)v;
  }
}

// ---------------------------------------------------------------------------
// Pack B (row-major KxN f32, leading dim ldb) into WMMA 32x16 f16 B fragments.
//   lane<16 : col=lane,    half h -> k = h
//   lane>=16: col=lane-16, half h -> k = 16 + h
// dst index: ((ntile*Kt + ktile)*32 + lane)*16 + h.  Zero-pads k>=K.
// ---------------------------------------------------------------------------
__global__ void pack_b_f16(const float* __restrict__ src, int ldb, int K, int N,
                           int Kt, _Float16* __restrict__ dst, int total) {
  int tid = blockIdx.x * blockDim.x + threadIdx.x;
  if (tid >= total) return;
  int lane  = tid & 31;
  int kt    = (tid >> 5) % Kt;
  int nt    = tid / (Kt * 32);
  int col   = nt * 16 + (lane & 15);
  int kbase = kt * 32 + ((lane >= 16) ? 16 : 0);
  _Float16* out = dst + ((size_t)(nt * Kt + kt) * 32 + lane) * 16;
#pragma unroll
  for (int h = 0; h < 16; ++h) {
    int k = kbase + h;
    float v = (k < K && col < N) ? src[(size_t)k * ldb + col] : 0.0f;
    out[h] = (_Float16)v;
  }
}

// ---------------------------------------------------------------------------
// WMMA GEMM: each wave computes a (16*MT) x (16*NT) block of C (f32).
// Requires Kt == 1 or Kt even (true for all call sites).
// ---------------------------------------------------------------------------
template <int MT, int NT>
__global__ __launch_bounds__(128, 1)
void wmma_gemm_f16(const _Float16* __restrict__ Apk,
                   const _Float16* __restrict__ Bpk,
                   float* __restrict__ C,
                   int N, int Kt, int Ngroups, int G) {
  int wid  = (int)((blockIdx.x * blockDim.x + threadIdx.x) >> 5);
  int lane = threadIdx.x & 31;
  if (wid >= G) return;
  int mg = wid / Ngroups;
  int ng = wid % Ngroups;

  const v16h* Af = (const v16h*)Apk + (size_t)(mg * MT) * Kt * 32 + lane;
  const v16h* Bf = (const v16h*)Bpk + (size_t)(ng * NT) * Kt * 32 + lane;

  v8f zero = {0.f, 0.f, 0.f, 0.f, 0.f, 0.f, 0.f, 0.f};
  v8f acc[MT][NT];
#pragma unroll
  for (int i = 0; i < MT; ++i)
#pragma unroll
    for (int j = 0; j < NT; ++j) acc[i][j] = zero;

  v16h a0[MT], b0[NT], a1[MT], b1[NT];

#define LOAD_FRAGS(AV, BV, KT)                                   \
  do {                                                           \
    _Pragma("unroll")                                            \
    for (int i = 0; i < MT; ++i)                                 \
      (AV)[i] = Af[((size_t)i * Kt + (KT)) * 32];                \
    _Pragma("unroll")                                            \
    for (int j = 0; j < NT; ++j)                                 \
      (BV)[j] = Bf[((size_t)j * Kt + (KT)) * 32];                \
  } while (0)

#define DO_MMAS(AV, BV)                                          \
  do {                                                           \
    _Pragma("unroll")                                            \
    for (int i = 0; i < MT; ++i)                                 \
      _Pragma("unroll")                                          \
      for (int j = 0; j < NT; ++j)                               \
        acc[i][j] = __builtin_amdgcn_wmma_f32_16x16x32_f16(      \
            false, (AV)[i], false, (BV)[j], (short)0,            \
            acc[i][j], false, false);                            \
  } while (0)

  if (Kt == 1) {
    LOAD_FRAGS(a0, b0, 0);
    DO_MMAS(a0, b0);
  } else {
    // Kt even: exit-free ping-pong software pipeline.
    LOAD_FRAGS(a0, b0, 0);
    LOAD_FRAGS(a1, b1, 1);
    int kt = 0;
    for (; kt + 2 < Kt; kt += 2) {
      DO_MMAS(a0, b0);
      LOAD_FRAGS(a0, b0, kt + 2);
      DO_MMAS(a1, b1);
      LOAD_FRAGS(a1, b1, kt + 3);
    }
    DO_MMAS(a0, b0);
    DO_MMAS(a1, b1);
  }
#undef LOAD_FRAGS
#undef DO_MMAS

  // C/D layout: VGPR r: lanes 0-15 -> (M=r, N=lane); lanes 16-31 -> (M=8+r).
  int lr = lane & 15;
  int hb = (lane >> 4) * 8;
#pragma unroll
  for (int i = 0; i < MT; ++i) {
#pragma unroll
    for (int j = 0; j < NT; ++j) {
      int col = (ng * NT + j) * 16 + lr;
      size_t base = (size_t)((mg * MT + i) * 16 + hb) * N + col;
#pragma unroll
      for (int r = 0; r < 8; ++r) C[base + (size_t)r * N] = acc[i][j][r];
    }
  }
}

// ---------------------------------------------------------------------------
// Causal depthwise conv1d (k=4, left pad 3) + bias + SiLU.
// xc = xz[:, 0:DINNER]; out xs (BL, DINNER).
// ---------------------------------------------------------------------------
__global__ void conv_silu(const float* __restrict__ xz,
                          const float* __restrict__ cw,   // (DINNER, 1, 4)
                          const float* __restrict__ cb,   // (DINNER)
                          float* __restrict__ xs, int total) {
  int tid = blockIdx.x * blockDim.x + threadIdx.x;
  if (tid >= total) return;
  int d = tid & (DINNER - 1);
  int t = (tid >> 11) & (L_DIM - 1);
  int b = tid >> 22;
  float acc = cb[d];
#pragma unroll
  for (int j = 0; j < 4; ++j) {
    int tt = t + j - 3;
    if (tt >= 0)
      acc += cw[d * 4 + j] * xz[((size_t)(b * L_DIM + tt) * (2 * DINNER)) + d];
  }
  xs[tid] = acc / (1.0f + __expf(-acc));
}

// ---------------------------------------------------------------------------
// Selective scan. One lane per (b, d, n) state; 16 consecutive lanes form one
// channel, reduced with shfl_xor butterflies each step. Lane n==0 applies the
// skip term D*xs and the SiLU(z) gate and writes y_gated.
// ---------------------------------------------------------------------------
__global__ void s6_scan(const float* __restrict__ dp,    // delta_pre (BL, DINNER)
                        const float* __restrict__ dbc,   // (BL, 32): [d_raw | B]
                        const float* __restrict__ xs,    // (BL, DINNER)
                        const float* __restrict__ xz,    // (BL, 2*DINNER)
                        const float* __restrict__ Alog,  // (DINNER, 16)
                        const float* __restrict__ Dp,    // (DINNER)
                        const float* __restrict__ bdt,   // (DINNER)
                        float* __restrict__ yg) {        // (BL, DINNER)
  int tid = blockIdx.x * blockDim.x + threadIdx.x;       // 65536 total
  int n  = tid & 15;
  int ch = tid >> 4;
  int d  = ch & (DINNER - 1);
  int b  = ch >> 11;

  float A  = -__expf(Alog[d * DSTATE + n]);
  float bd = bdt[d];
  float Dd = Dp[d];
  float h  = 0.0f;

  for (int t = 0; t < L_DIM; ++t) {
    size_t row = (size_t)(b * L_DIM + t);
    float dpre  = dp[row * DINNER + d] + bd;
    float delta = (dpre > 20.0f) ? dpre : __logf(1.0f + __expf(dpre));
    float Bn = dbc[row * 32 + DSTATE + n];               // B (and C == B)
    float xv = xs[row * DINNER + d];
    h = __expf(delta * A) * h + delta * Bn * xv;
    float y = h * Bn;                                    // h * C[n]
    y += __shfl_xor(y, 1);
    y += __shfl_xor(y, 2);
    y += __shfl_xor(y, 4);
    y += __shfl_xor(y, 8);
    if (n == 0) {
      float zv = xz[row * (2 * DINNER) + DINNER + d];
      float out = (y + Dd * xv) * (zv / (1.0f + __expf(-zv)));
      yg[row * DINNER + d] = out;
    }
  }
}

// ---------------------------------------------------------------------------
extern "C" void kernel_launch(void* const* d_in, const int* in_sizes, int n_in,
                              void* d_out, int out_size, void* d_ws, size_t ws_size,
                              hipStream_t stream) {
  const float* x      = (const float*)d_in[0];
  const float* W_in   = (const float*)d_in[1];
  const float* conv_w = (const float*)d_in[2];
  const float* conv_b = (const float*)d_in[3];
  const float* W_x    = (const float*)d_in[4];
  const float* W_dt   = (const float*)d_in[5];
  const float* b_dt   = (const float*)d_in[6];
  const float* A_log  = (const float*)d_in[7];
  const float* D_p    = (const float*)d_in[8];
  const float* W_out  = (const float*)d_in[9];

  char* ws = (char*)d_ws;
  float*     xz  = (float*)(ws + OFF_XZ);
  float*     xs  = (float*)(ws + OFF_XS);
  float*     dbc = (float*)(ws + OFF_DBC);
  float*     dp  = (float*)(ws + OFF_DP);
  float*     yg  = (float*)(ws + OFF_YG);
  _Float16*  Apk = (_Float16*)(ws + OFF_APK);
  _Float16*  Bpk = (_Float16*)(ws + OFF_BPK);

  const int Mt = BL / 16;  // 256 M-tiles everywhere

  // ---- 1. xz = x @ W_in   (M=4096, K=1024, N=4096) ----
  {
    int Kt = DMODEL / 32, Ntl = (2 * DINNER) / 16;      // 32, 256
    int ta = Mt * Kt * 32, tb = Ntl * Kt * 32;
    pack_a_f16<<<(ta + 255) / 256, 256, 0, stream>>>(x, DMODEL, BL, DMODEL, Kt, Apk, ta);
    pack_b_f16<<<(tb + 255) / 256, 256, 0, stream>>>(W_in, 2 * DINNER, DMODEL, 2 * DINNER, Kt, Bpk, tb);
    int Ng = Ntl / 4, G = (Mt / 2) * Ng;                // 128*64 = 8192 waves
    wmma_gemm_f16<2, 4><<<(G * 32 + 127) / 128, 128, 0, stream>>>(Apk, Bpk, xz, 2 * DINNER, Kt, Ng, G);
  }

  // ---- 2. xs = silu(conv(xc) + b) ----
  {
    int total = BL * DINNER;
    conv_silu<<<(total + 255) / 256, 256, 0, stream>>>(xz, conv_w, conv_b, xs, total);
  }

  // ---- 3. dBC = xs @ W_x  (M=4096, K=2048, N=32) ----
  {
    int Kt = DINNER / 32, Ntl = (2 * DSTATE) / 16;      // 64, 2
    int ta = Mt * Kt * 32, tb = Ntl * Kt * 32;
    pack_a_f16<<<(ta + 255) / 256, 256, 0, stream>>>(xs, DINNER, BL, DINNER, Kt, Apk, ta);
    pack_b_f16<<<(tb + 255) / 256, 256, 0, stream>>>(W_x, 2 * DSTATE, DINNER, 2 * DSTATE, Kt, Bpk, tb);
    int Ng = Ntl / 2, G = Mt * Ng;                      // 256 waves, keep parallelism
    wmma_gemm_f16<1, 2><<<(G * 32 + 127) / 128, 128, 0, stream>>>(Apk, Bpk, dbc, 2 * DSTATE, Kt, Ng, G);
  }

  // ---- 4. delta_pre = d_raw @ W_dt  (M=4096, K=16 (pad 32), N=2048) ----
  {
    int Kt = 1, Ntl = DINNER / 16;                      // 1, 128
    int ta = Mt * Kt * 32, tb = Ntl * Kt * 32;
    pack_a_f16<<<(ta + 255) / 256, 256, 0, stream>>>(dbc, 2 * DSTATE, BL, DSTATE, Kt, Apk, ta);
    pack_b_f16<<<(tb + 255) / 256, 256, 0, stream>>>(W_dt, DINNER, DSTATE, DINNER, Kt, Bpk, tb);
    int Ng = Ntl / 4, G = (Mt / 2) * Ng;                // 128*32 = 4096 waves
    wmma_gemm_f16<2, 4><<<(G * 32 + 127) / 128, 128, 0, stream>>>(Apk, Bpk, dp, DINNER, Kt, Ng, G);
  }

  // ---- 5. selective scan + skip + gate ----
  {
    int total = B_DIM * DINNER * DSTATE;                // 65536
    s6_scan<<<(total + 255) / 256, 256, 0, stream>>>(dp, dbc, xs, xz, A_log, D_p, b_dt, yg);
  }

  // ---- 6. out = y_gated @ W_out  (M=4096, K=2048, N=1024) -> d_out ----
  {
    int Kt = DINNER / 32, Ntl = DMODEL / 16;            // 64, 64
    int ta = Mt * Kt * 32, tb = Ntl * Kt * 32;
    pack_a_f16<<<(ta + 255) / 256, 256, 0, stream>>>(yg, DINNER, BL, DINNER, Kt, Apk, ta);
    pack_b_f16<<<(tb + 255) / 256, 256, 0, stream>>>(W_out, DMODEL, DINNER, DMODEL, Kt, Bpk, tb);
    int Ng = Ntl / 4, G = (Mt / 2) * Ng;                // 128*16 = 2048 waves
    wmma_gemm_f16<2, 4><<<(G * 32 + 127) / 128, 128, 0, stream>>>(Apk, Bpk, (float*)d_out, DMODEL, Kt, Ng, G);
  }
}
